// PillarsScatter_34634616275490
// MI455X (gfx1250) — matrix operand order repaired and checked by reading, not compiled, not run on hardware
//
#include <hip/hip_runtime.h>

typedef __attribute__((ext_vector_type(16))) _Float16 v16h;
typedef __attribute__((ext_vector_type(8)))  _Float16 v8h;
typedef __attribute__((ext_vector_type(8)))  float    v8f;
typedef __attribute__((ext_vector_type(4)))  float    v4f;

namespace {
constexpr int kB     = 2;
constexpr int kC     = 64;
constexpr int kNZ    = 16;
constexpr int kPlane = 256 * 256;               // NX*NY = 65536 rows per batch
constexpr int kRows  = kB * kPlane;             // 131072 M-rows total
constexpr long long kFvElems = (long long)kRows * kC;   // 8388608 floats in ws
constexpr int kOutHalf = kB * kNZ * kPlane;     // 2097152 floats per output tensor
constexpr float kLnEps = 1e-5f;
}

// ---------------------------------------------------------------- init ------
__global__ __launch_bounds__(256) void pv_init_kernel(float* __restrict__ fv,
                                                      float* __restrict__ out_binary) {
  long long i = (long long)blockIdx.x * blockDim.x + threadIdx.x;
  float4 z = make_float4(0.f, 0.f, 0.f, 0.f);
  if (i < kFvElems / 4) ((float4*)fv)[i] = z;         // zero z-summed canvas
  if (i < kOutHalf / 4) ((float4*)out_binary)[i] = z; // zero binary canvas
}

// -------------------------------------------------------------- scatter -----
// 64 lanes per voxel: coalesced read of the 64 channels, atomic add into the
// z-collapsed accumulator (canvas.sum(axis=1) == add over z).
__global__ __launch_bounds__(256) void pv_scatter_kernel(
    const float* __restrict__ feat, const int* __restrict__ coors,
    float* __restrict__ fv, float* __restrict__ out_binary, int n) {
  int t = blockIdx.x * blockDim.x + threadIdx.x;
  int v = t >> 6;
  int c = t & 63;
  if (v >= n) return;
  int4 cr = ((const int4*)coors)[v];
  int b = cr.x;
  int z = min(cr.y, kNZ - 1);
  int x = min(cr.z, 255);
  int y = min(cr.w, 255);
  float f = feat[(long long)v * kC + c];
  long long row = ((long long)b * 256 + x) * 256 + y;
  atomicAdd(&fv[row * kC + c], f);
  if (c == 0) out_binary[((b * kNZ + z) * 256 + x) * 256 + y] = 1.0f;
}

// ------------------------------------------------------------ fused MLP -----
// 16-lane row-sum on the VALU via DPP16 (no LDS pipe traffic). The butterfly
// quad_perm(xor1), quad_perm(xor2), row_half_mirror (xor7 inside 8),
// row_mirror (xor15 inside 16) never crosses a 16-lane row, so lanes 0-15 and
// 16-31 (the two D-matrix row groups) reduce independently.
template <int CTRL>
__device__ __forceinline__ float dpp_add(float x) {
  int yi = __builtin_amdgcn_update_dpp(0, __float_as_int(x), CTRL, 0xF, 0xF, true);
  return x + __int_as_float(yi);
}

__device__ __forceinline__ float half16_sum(float v) {
  v = dpp_add<0x0B1>(v);  // quad_perm [1,0,3,2]  (xor 1)
  v = dpp_add<0x04E>(v);  // quad_perm [2,3,0,1]  (xor 2)
  v = dpp_add<0x141>(v);  // row_half_mirror      (xor 7 within 8)
  v = dpp_add<0x140>(v);  // row_mirror           (xor 15 within 16)
  return v;
}

__global__ __launch_bounds__(256) void pv_mlp_kernel(
    const float* __restrict__ fv, const float* __restrict__ nz_embed,
    const float* __restrict__ W1, const float* __restrict__ b1,
    const float* __restrict__ g1, const float* __restrict__ be1,
    const float* __restrict__ W2, const float* __restrict__ b2,
    const float* __restrict__ g2, const float* __restrict__ be2,
    float* __restrict__ out) {
  // row stride 40 halfs (80B): 16B-aligned v8h reads, banks 20*r mod 64 are
  // distinct for r=0..15 -> conflict-free ds_load_b128 for the A2 fragment.
  __shared__ _Float16 lds_h[8][16][40];  // per-wave relu(LN(h1)) tile, f16
  __shared__ float    lds_o[8][16][17];  // per-wave final tile for transpose

  const int lane = threadIdx.x & 31;
  const int w    = threadIdx.x >> 5;
  const int tile = blockIdx.x * 8 + w;   // 16 M-rows per wave
  const int row  = lane & 15;            // M (A frag) / N column (B, C/D frags)
  const int hi   = lane >> 4;

  // nz_embed.sum() broadcast to all lanes
  float nzv = (lane < 16) ? nz_embed[lane] : 0.0f;
  nzv = half16_sum(nzv);
  nzv += __shfl_xor(nzv, 16);
  const float nz_sum = nzv;

  // ---- B1 fragments (W1: 64x16 row-major), two K-chunks of 32 ----
  const int col = row;
  const int kb  = hi * 8;
  v16h bf0, bf1;
#pragma unroll
  for (int j = 0; j < 16; ++j) {
    bf0[j] = (_Float16)W1[(kb + j + (j >= 8 ? 8 : 0)) * 16 + col];
    bf1[j] = (_Float16)W1[(32 + kb + j + (j >= 8 ? 8 : 0)) * 16 + col];
  }

  // ---- A fragments: 16 rows x 64 K of f32, + nz_sum, converted to f16 ----
  const float* arow = fv + (long long)(tile * 16 + row) * kC;
  __builtin_prefetch(fv + (long long)((tile + 8) * 16 + row) * kC, 0, 3);
  const int koff = 8 * hi;
  v16h a0, a1;
#pragma unroll
  for (int ch = 0; ch < 2; ++ch) {
    const float* p = arow + ch * 32;
    v4f q0 = *(const v4f*)(p + koff);
    v4f q1 = *(const v4f*)(p + koff + 4);
    v4f q2 = *(const v4f*)(p + 16 + koff);
    v4f q3 = *(const v4f*)(p + 16 + koff + 4);
    v16h* a = ch ? &a1 : &a0;
#pragma unroll
    for (int i = 0; i < 4; ++i) {
      (*a)[i]      = (_Float16)(q0[i] + nz_sum);
      (*a)[4 + i]  = (_Float16)(q1[i] + nz_sum);
      (*a)[8 + i]  = (_Float16)(q2[i] + nz_sum);
      (*a)[12 + i] = (_Float16)(q3[i] + nz_sum);
    }
  }

  // ---- GEMM1: h = fvtile @ W1 (K=64 -> 2 WMMAs, f32 accumulate) ----
  v8f acc = {};
  acc = __builtin_amdgcn_wmma_f32_16x16x32_f16(false, a0, false, bf0,
                                               (short)0, acc, false, false);
  acc = __builtin_amdgcn_wmma_f32_16x16x32_f16(false, a1, false, bf1,
                                               (short)0, acc, false, false);

  // ---- bias + LayerNorm(16) + ReLU; D row M lives across 16 lanes ----
  const float bias1 = b1[col], gg1 = g1[col], bb1 = be1[col];
#pragma unroll
  for (int r = 0; r < 8; ++r) {
    float x  = acc[r] + bias1;
    float s  = half16_sum(x) * (1.0f / 16.0f);
    float s2 = half16_sum(x * x) * (1.0f / 16.0f);
    float var = s2 - s * s;
    float xn  = (x - s) * rsqrtf(var + kLnEps) * gg1 + bb1;
    xn = fmaxf(xn, 0.0f);
    lds_h[w][r + 8 * hi][col] = (_Float16)xn;
  }
  __builtin_amdgcn_wave_barrier();  // wave-private LDS: DS ops are in-order

  // ---- A2 fragment from LDS (K=16, padded to 32 with zeros) ----
  v16h a2;
  v8h av = *(const v8h*)&lds_h[w][row][koff];
#pragma unroll
  for (int j = 0; j < 8; ++j) { a2[j] = av[j]; a2[8 + j] = (_Float16)0.0f; }

  // ---- B2 fragment (W2: 16x16), lanes with K>=16 are zero padding ----
  v16h b2f;
#pragma unroll
  for (int j = 0; j < 16; ++j) {
    float wv = (j < 8) ? W2[(kb + j) * 16 + col] : 0.0f;
    b2f[j] = (_Float16)wv;
  }

  // ---- GEMM2 + bias + LayerNorm ----
  v8f acc2 = {};
  acc2 = __builtin_amdgcn_wmma_f32_16x16x32_f16(false, a2, false, b2f,
                                                (short)0, acc2, false, false);
  const float bias2 = b2[col], gg2 = g2[col], bb2 = be2[col];
#pragma unroll
  for (int r = 0; r < 8; ++r) {
    float x  = acc2[r] + bias2;
    float s  = half16_sum(x) * (1.0f / 16.0f);
    float s2 = half16_sum(x * x) * (1.0f / 16.0f);
    float var = s2 - s * s;
    float xn  = (x - s) * rsqrtf(var + kLnEps) * gg2 + bb2;
    lds_o[w][r + 8 * hi][col] = xn;
  }
  __builtin_amdgcn_wave_barrier();

  // ---- transpose-store: out[b, n, xy] = h2[xy, n]; 64B-contiguous rows ----
  const int rowg0 = tile * 16;
  const int bidx  = rowg0 >> 16;       // kPlane rows per batch
  const int xyb   = rowg0 & (kPlane - 1);
  float* obase = out + (long long)bidx * kNZ * kPlane + xyb;
#pragma unroll
  for (int i = 0; i < 8; ++i) {
    int idx = i * 32 + lane;
    int nn = idx >> 4;
    int mm = idx & 15;
    obase[(long long)nn * kPlane + mm] = lds_o[w][mm][nn];
  }
}

// --------------------------------------------------------------- launch -----
extern "C" void kernel_launch(void* const* d_in, const int* in_sizes, int n_in,
                              void* d_out, int out_size, void* d_ws, size_t ws_size,
                              hipStream_t stream) {
  const float* feat = (const float*)d_in[0];
  const int*   coor = (const int*)d_in[1];
  const float* nze  = (const float*)d_in[2];
  const float* W1   = (const float*)d_in[3];
  const float* b1   = (const float*)d_in[4];
  const float* g1   = (const float*)d_in[5];
  const float* be1  = (const float*)d_in[6];
  const float* W2   = (const float*)d_in[7];
  const float* b2   = (const float*)d_in[8];
  const float* g2   = (const float*)d_in[9];
  const float* be2  = (const float*)d_in[10];

  float* out_batch  = (float*)d_out;                 // (B,16,256,256)
  float* out_binary = (float*)d_out + kOutHalf;      // (B,16,256,256)
  float* fv         = (float*)d_ws;                  // (B*65536, 64) accumulator

  // 1) zero accumulator + binary canvas (fv/4 = 2097152 float4 -> 8192 blocks)
  pv_init_kernel<<<(int)(kFvElems / 4 / 256), 256, 0, stream>>>(fv, out_binary);

  // 2) scatter-add over z (64 lanes per voxel)
  int n = in_sizes[0] / kC;  // 100000
  int sb = (n * 64 + 255) / 256;
  pv_scatter_kernel<<<sb, 256, 0, stream>>>(feat, coor, fv, out_binary, n);

  // 3) fused GEMM1+LN+ReLU+GEMM2+LN+transpose: 8192 tiles, 8 waves/block
  pv_mlp_kernel<<<kRows / 16 / 8, 256, 0, stream>>>(
      fv, nze, W1, b1, g1, be1, W2, b2, g2, be2, out_batch);
}
